// RegularizedDevignModel_45483703665341
// MI455X (gfx1250) — matrix operand order repaired and checked by reading, not compiled, not run on hardware
//
#include <hip/hip_runtime.h>
#include <math.h>

// ---------------- problem constants ----------------
#define NNODES   100000
#define NEDGES   1600000
#define NGRAPHS  512
#define HID      200
#define IN_DIM   205
#define HID3     600
#define HP       208     // HID padded to mult of 16 (C stride)
#define HPB      256     // HID padded to mult of 64 (Bt rows)
#define H3P      608     // 3*HID padded to mult of 16 (C stride)
#define H3PB     640     // 3*HID padded to mult of 64 (Bt rows)
#define KPH      224     // K=200/205 padded to mult of 32
#define K2       400     // fc1 K
#define KP2      416     // 400 padded to mult of 32
#define EPSBN    1e-5f

// ---------------- vector types for WMMA ----------------
typedef unsigned short ushort8_t  __attribute__((ext_vector_type(8)));
typedef __bf16         bf16x16    __attribute__((ext_vector_type(16)));
typedef float          floatx8    __attribute__((ext_vector_type(8)));

struct U8Pair { ushort8_t lo, hi; };   // 32 bytes == bf16x16

__device__ __forceinline__ unsigned short f32_to_bf16(float f) {
  unsigned int u = __float_as_uint(f);
  unsigned int r = u + 0x7FFFu + ((u >> 16) & 1u);       // RNE
  if ((u & 0x7F800000u) == 0x7F800000u) r = u;           // inf/nan passthrough
  return (unsigned short)(r >> 16);
}

__device__ __forceinline__ float sigmoidf_(float x) { return 1.f / (1.f + __expf(-x)); }

// ---------------- weight packing: Bt[n][k] (bf16, zero padded) ----------------
__global__ __launch_bounds__(256) void pack_bt(const float* __restrict__ W,
                                               unsigned short* __restrict__ Bt,
                                               int Nn, int Kk, int Np, int Kp, int transpose) {
  int t = blockIdx.x * blockDim.x + threadIdx.x;
  if (t >= Np * Kp) return;
  int n = t / Kp, k = t - n * Kp;
  float v = 0.f;
  if (n < Nn && k < Kk) v = transpose ? W[(size_t)k * Nn + n] : W[(size_t)n * Kk + k];
  Bt[t] = f32_to_bf16(v);
}

// ---------------- activation packing: fp32 [M][K] -> bf16 [M][Kp] ----------------
__global__ __launch_bounds__(256) void pack_a(const float* __restrict__ S,
                                              unsigned short* __restrict__ D,
                                              int M, int K, int Kp) {
  int t = blockIdx.x * blockDim.x + threadIdx.x;
  if (t >= M * Kp) return;
  int m = t / Kp, k = t - m * Kp;
  float v = (k < K) ? S[(size_t)m * K + k] : 0.f;
  D[t] = f32_to_bf16(v);
}

// ---------------- blocked WMMA GEMM ----------------
// C[M][Np] = A[M][Kp] * Bt[NpB][Kp]^T  with NpB = 64-aligned >= Np (zero pad rows).
// One wave computes a 32x64 macro-tile: 2 M-tiles x 4 N-tiles, 8 independent
// accumulator chains -> 8 v_wmma per k-step against 4 A-loads + 8 B-loads.
__global__ __launch_bounds__(256) void wmma_gemm_blk(const unsigned short* __restrict__ A,
                                                     const unsigned short* __restrict__ Bt,
                                                     float* __restrict__ C,
                                                     int Mgroups, int Ngroups, int Kp, int Np) {
  int wid = (blockIdx.x * (int)blockDim.x + (int)threadIdx.x) >> 5;
  if (wid >= Mgroups * Ngroups) return;
  int lane = threadIdx.x & 31;
  int mg = wid / Ngroups;
  int ng = wid - mg * Ngroups;
  int hi  = lane >> 4;     // half-wave select
  int l15 = lane & 15;

  const unsigned short* arow0 = A + (size_t)(mg * 32 + l15) * Kp;
  const unsigned short* arow1 = arow0 + (size_t)16 * Kp;
  const unsigned short* brow0 = Bt + (size_t)(ng * 64 + l15) * Kp;
  const unsigned short* brow1 = brow0 + (size_t)16 * Kp;
  const unsigned short* brow2 = brow0 + (size_t)32 * Kp;
  const unsigned short* brow3 = brow0 + (size_t)48 * Kp;

  floatx8 acc00 = {}, acc01 = {}, acc02 = {}, acc03 = {};
  floatx8 acc10 = {}, acc11 = {}, acc12 = {}, acc13 = {};

  for (int k0 = 0; k0 < Kp; k0 += 32) {
    int ka = k0 + 16 * hi;       // A chunk offset : K 0..15 (lo half) / 16..31 (hi half)
    int kb = k0 + 16 * hi;       // B chunk offset : K 0..15 (lo half) / 16..31 (hi half)
    U8Pair ap0, ap1, bp0, bp1, bp2, bp3;
    ap0.lo = *(const ushort8_t*)(arow0 + ka);
    ap0.hi = *(const ushort8_t*)(arow0 + ka + 8);
    ap1.lo = *(const ushort8_t*)(arow1 + ka);
    ap1.hi = *(const ushort8_t*)(arow1 + ka + 8);
    bp0.lo = *(const ushort8_t*)(brow0 + kb);
    bp0.hi = *(const ushort8_t*)(brow0 + kb + 8);
    bp1.lo = *(const ushort8_t*)(brow1 + kb);
    bp1.hi = *(const ushort8_t*)(brow1 + kb + 8);
    bp2.lo = *(const ushort8_t*)(brow2 + kb);
    bp2.hi = *(const ushort8_t*)(brow2 + kb + 8);
    bp3.lo = *(const ushort8_t*)(brow3 + kb);
    bp3.hi = *(const ushort8_t*)(brow3 + kb + 8);
    bf16x16 av0 = __builtin_bit_cast(bf16x16, ap0);
    bf16x16 av1 = __builtin_bit_cast(bf16x16, ap1);
    bf16x16 bv0 = __builtin_bit_cast(bf16x16, bp0);
    bf16x16 bv1 = __builtin_bit_cast(bf16x16, bp1);
    bf16x16 bv2 = __builtin_bit_cast(bf16x16, bp2);
    bf16x16 bv3 = __builtin_bit_cast(bf16x16, bp3);
    acc00 = __builtin_amdgcn_wmma_f32_16x16x32_bf16(false, av0, false, bv0, (short)0, acc00, false, false);
    acc01 = __builtin_amdgcn_wmma_f32_16x16x32_bf16(false, av0, false, bv1, (short)0, acc01, false, false);
    acc02 = __builtin_amdgcn_wmma_f32_16x16x32_bf16(false, av0, false, bv2, (short)0, acc02, false, false);
    acc03 = __builtin_amdgcn_wmma_f32_16x16x32_bf16(false, av0, false, bv3, (short)0, acc03, false, false);
    acc10 = __builtin_amdgcn_wmma_f32_16x16x32_bf16(false, av1, false, bv0, (short)0, acc10, false, false);
    acc11 = __builtin_amdgcn_wmma_f32_16x16x32_bf16(false, av1, false, bv1, (short)0, acc11, false, false);
    acc12 = __builtin_amdgcn_wmma_f32_16x16x32_bf16(false, av1, false, bv2, (short)0, acc12, false, false);
    acc13 = __builtin_amdgcn_wmma_f32_16x16x32_bf16(false, av1, false, bv3, (short)0, acc13, false, false);
  }

  // C layout per tile: VGPR v -> row +v + 8*hi, col = l15
  floatx8 accs[2][4] = {{acc00, acc01, acc02, acc03}, {acc10, acc11, acc12, acc13}};
#pragma unroll
  for (int i = 0; i < 2; ++i) {
#pragma unroll
    for (int j = 0; j < 4; ++j) {
      int ncol = (ng * 4 + j) * 16;
      if (ncol < Np) {                               // padded N tiles are discarded
        float* cbase = C + (size_t)(mg * 32 + i * 16 + 8 * hi) * Np + ncol + l15;
#pragma unroll
        for (int v = 0; v < 8; ++v) cbase[(size_t)v * Np] = accs[i][j][v];
      }
    }
  }
}

// ---------------- proj epilogue: bias + BN1 + ReLU -> h fp32 + h bf16(padded) ----------------
__global__ __launch_bounds__(256) void epi_bn_relu_pack(const float* __restrict__ C,
                                                        const float* __restrict__ bias,
                                                        const float* __restrict__ g,
                                                        const float* __restrict__ b,
                                                        const float* __restrict__ mm,
                                                        const float* __restrict__ vv,
                                                        float* __restrict__ H,
                                                        unsigned short* __restrict__ Hb,
                                                        int M) {
  int t = blockIdx.x * blockDim.x + threadIdx.x;
  if (t >= M * KPH) return;
  int row = t / KPH, c = t - row * KPH;
  float val = 0.f;
  if (c < HID) {
    val = C[(size_t)row * HP + c] + bias[c];
    val = (val - mm[c]) * (g[c] * rsqrtf(vv[c] + EPSBN)) + b[c];
    val = fmaxf(val, 0.f);
    H[(size_t)row * HID + c] = val;
  }
  Hb[(size_t)row * KPH + c] = f32_to_bf16(val);
}

// ---------------- edge scatter-add: agg[dst] += m[src]  (4 channels / thread) ----------------
__global__ __launch_bounds__(256) void scatter_add(const float* __restrict__ Mm,
                                                   const int* __restrict__ src,
                                                   const int* __restrict__ dst,
                                                   float* __restrict__ agg, int total) {
  int t = blockIdx.x * blockDim.x + threadIdx.x;
  if (t >= total) return;
  int e = t / 50;
  int c = (t - e * 50) * 4;
  int s = src[e], d = dst[e];
  const float4 mv = *(const float4*)(Mm + (size_t)s * HP + c);
  float* ap = agg + (size_t)d * HID + c;
  atomicAdd(ap + 0, mv.x);
  atomicAdd(ap + 1, mv.y);
  atomicAdd(ap + 2, mv.z);
  atomicAdd(ap + 3, mv.w);
}

// ---------------- GRU cell elementwise ----------------
__global__ __launch_bounds__(256) void gru_elementwise(const float* __restrict__ GI,
                                                       const float* __restrict__ GH,
                                                       const float* __restrict__ bih,
                                                       const float* __restrict__ bhh,
                                                       float* __restrict__ H,
                                                       unsigned short* __restrict__ Hb,
                                                       int M) {
  int t = blockIdx.x * blockDim.x + threadIdx.x;
  if (t >= M * KPH) return;
  int row = t / KPH, c = t - row * KPH;
  float hnew = 0.f;
  if (c < HID) {
    size_t g0 = (size_t)row * H3P;
    float ir  = GI[g0 + c]           + bih[c];
    float iz  = GI[g0 + HID + c]     + bih[HID + c];
    float inn = GI[g0 + 2 * HID + c] + bih[2 * HID + c];
    float hr  = GH[g0 + c]           + bhh[c];
    float hz  = GH[g0 + HID + c]     + bhh[HID + c];
    float hn  = GH[g0 + 2 * HID + c] + bhh[2 * HID + c];
    float r = sigmoidf_(ir + hr);
    float z = sigmoidf_(iz + hz);
    float nn = tanhf(inn + r * hn);
    float hold = H[(size_t)row * HID + c];
    hnew = (1.f - z) * nn + z * hold;
    H[(size_t)row * HID + c] = hnew;
  }
  Hb[(size_t)row * KPH + c] = f32_to_bf16(hnew);
}

// ---------------- BN2 + ReLU, in-place on h ----------------
__global__ __launch_bounds__(256) void bn_relu_inplace(float* __restrict__ H,
                                                       const float* __restrict__ g,
                                                       const float* __restrict__ b,
                                                       const float* __restrict__ mm,
                                                       const float* __restrict__ vv,
                                                       int M) {
  int t = blockIdx.x * blockDim.x + threadIdx.x;
  if (t >= M * HID) return;
  int c = t % HID;
  float val = (H[t] - mm[c]) * (g[c] * rsqrtf(vv[c] + EPSBN)) + b[c];
  H[t] = fmaxf(val, 0.f);
}

// ---------------- pooling ----------------
__global__ __launch_bounds__(256) void pool_count(const int* __restrict__ batch,
                                                  float* __restrict__ cnt, int M) {
  int t = blockIdx.x * blockDim.x + threadIdx.x;
  if (t >= M) return;
  atomicAdd(&cnt[batch[t]], 1.f);
}

__global__ __launch_bounds__(256) void pool_main(const float* __restrict__ H,
                                                 const int* __restrict__ batch,
                                                 float* __restrict__ sum,
                                                 unsigned int* __restrict__ maxb, int M) {
  int t = blockIdx.x * blockDim.x + threadIdx.x;
  if (t >= M * HID) return;
  int node = t / HID, c = t - node * HID;
  float val = H[t];                       // >= 0 post-ReLU, so uint-compare == float-compare
  int gph = batch[node];
  atomicAdd(&sum[(size_t)gph * HID + c], val);
  atomicMax(&maxb[(size_t)gph * HID + c], __float_as_uint(val));
}

__global__ __launch_bounds__(256) void pool_finalize(const float* __restrict__ sum,
                                                     const unsigned int* __restrict__ maxb,
                                                     const float* __restrict__ cnt,
                                                     unsigned short* __restrict__ Gb) {
  int t = blockIdx.x * blockDim.x + threadIdx.x;
  if (t >= NGRAPHS * KP2) return;
  int g = t / KP2, k = t - g * KP2;
  float val = 0.f;
  if (k < HID)            val = sum[(size_t)g * HID + k] / fmaxf(cnt[g], 1.f);
  else if (k < 2 * HID)   val = __uint_as_float(maxb[(size_t)g * HID + (k - HID)]);
  Gb[t] = f32_to_bf16(val);
}

// ---------------- fc1 epilogue: bias + BNf + ReLU ----------------
__global__ __launch_bounds__(256) void fc1_epilogue(const float* __restrict__ C,
                                                    const float* __restrict__ bias,
                                                    const float* __restrict__ g,
                                                    const float* __restrict__ b,
                                                    const float* __restrict__ mm,
                                                    const float* __restrict__ vv,
                                                    float* __restrict__ GG) {
  int t = blockIdx.x * blockDim.x + threadIdx.x;
  if (t >= NGRAPHS * HID) return;
  int row = t / HID, c = t - row * HID;
  float val = C[(size_t)row * HP + c] + bias[c];
  val = (val - mm[c]) * (g[c] * rsqrtf(vv[c] + EPSBN)) + b[c];
  GG[t] = fmaxf(val, 0.f);
}

// ---------------- fc2 (tiny) ----------------
__global__ __launch_bounds__(256) void fc2_kernel(const float* __restrict__ GG,
                                                  const float* __restrict__ W2,
                                                  const float* __restrict__ b2,
                                                  float* __restrict__ out) {
  int t = blockIdx.x * blockDim.x + threadIdx.x;
  if (t >= NGRAPHS * 2) return;
  int g = t / 2, o = t - g * 2;
  float acc = b2[o];
  for (int c = 0; c < HID; ++c) acc += GG[(size_t)g * HID + c] * W2[(size_t)o * HID + c];
  out[t] = acc;
}

// ================= host side =================
static inline int cdiv_i(long a, long b) { return (int)((a + b - 1) / b); }

extern "C" void kernel_launch(void* const* d_in, const int* in_sizes, int n_in,
                              void* d_out, int out_size, void* d_ws, size_t ws_size,
                              hipStream_t stream) {
  const float* x        = (const float*)d_in[0];
  const int*   ei       = (const int*)d_in[1];
  const int*   batch    = (const int*)d_in[2];
  const float* proj_W   = (const float*)d_in[3];
  const float* proj_b   = (const float*)d_in[4];
  const float* bn1_g    = (const float*)d_in[5];
  const float* bn1_b    = (const float*)d_in[6];
  const float* bn1_m    = (const float*)d_in[7];
  const float* bn1_v    = (const float*)d_in[8];
  const float* bn2_g    = (const float*)d_in[9];
  const float* bn2_b    = (const float*)d_in[10];
  const float* bn2_m    = (const float*)d_in[11];
  const float* bn2_v    = (const float*)d_in[12];
  const float* bnf_g    = (const float*)d_in[13];
  const float* bnf_b    = (const float*)d_in[14];
  const float* bnf_m    = (const float*)d_in[15];
  const float* bnf_v    = (const float*)d_in[16];
  const float* ggc_W    = (const float*)d_in[17];
  const float* gru_wih  = (const float*)d_in[18];
  const float* gru_whh  = (const float*)d_in[19];
  const float* gru_bih  = (const float*)d_in[20];
  const float* gru_bhh  = (const float*)d_in[21];
  const float* fc1_W    = (const float*)d_in[22];
  const float* fc1_b    = (const float*)d_in[23];
  const float* fc2_W    = (const float*)d_in[24];
  const float* fc2_b    = (const float*)d_in[25];
  float* out = (float*)d_out;

  const int* e_src = ei;            // edge_index[0]
  const int* e_dst = ei + NEDGES;   // edge_index[1]

  // ---- workspace carve-up (256B aligned) ----
  unsigned char* W = (unsigned char*)d_ws;
  size_t off = 0;
  auto take = [&](size_t bytes) { size_t o = off; off += (bytes + 255) & ~(size_t)255; return o; };
  unsigned short* bt_proj = (unsigned short*)(W + take((size_t)HPB  * KPH * 2));
  unsigned short* bt_ggc0 = (unsigned short*)(W + take((size_t)HPB  * KPH * 2));
  unsigned short* bt_ggc1 = (unsigned short*)(W + take((size_t)HPB  * KPH * 2));
  unsigned short* bt_ggc2 = (unsigned short*)(W + take((size_t)HPB  * KPH * 2));
  unsigned short* bt_wih  = (unsigned short*)(W + take((size_t)H3PB * KPH * 2));
  unsigned short* bt_whh  = (unsigned short*)(W + take((size_t)H3PB * KPH * 2));
  unsigned short* bt_fc1  = (unsigned short*)(W + take((size_t)HPB  * KP2 * 2));
  float*          h       = (float*)         (W + take((size_t)NNODES * HID * 4));
  unsigned short* h_bf    = (unsigned short*)(W + take((size_t)NNODES * KPH * 2));
  unsigned short* ax_aggb = (unsigned short*)(W + take((size_t)NNODES * KPH * 2)); // A_x, later agg_bf16
  float*          agg     = (float*)         (W + take((size_t)NNODES * HID * 4));
  float*          Cgi     = (float*)         (W + take((size_t)NNODES * H3P * 4));
  float*          R       = (float*)         (W + take((size_t)NNODES * H3P * 4)); // Cproj/Cm/Cgh
  float*          p_sum   = (float*)         (W + take((size_t)NGRAPHS * HID * 4));
  float*          p_cnt   = (float*)         (W + take((size_t)NGRAPHS * 4));
  unsigned int*   p_max   = (unsigned int*)  (W + take((size_t)NGRAPHS * HID * 4));
  unsigned short* g_bf    = (unsigned short*)(W + take((size_t)NGRAPHS * KP2 * 2));
  float*          Cfc1    = (float*)         (W + take((size_t)NGRAPHS * HP * 4));
  float*          GG      = (float*)         (W + take((size_t)NGRAPHS * HID * 4));
  (void)ws_size; (void)n_in; (void)in_sizes; (void)out_size;

  // ---- pack weights to bf16 Bt layout (N padded to 64-row groups) ----
  pack_bt<<<cdiv_i((long)HPB * KPH, 256), 256, 0, stream>>>(proj_W, bt_proj, HID, IN_DIM, HPB, KPH, 0);
  pack_bt<<<cdiv_i((long)HPB * KPH, 256), 256, 0, stream>>>(ggc_W + 0 * HID * HID, bt_ggc0, HID, HID, HPB, KPH, 1);
  pack_bt<<<cdiv_i((long)HPB * KPH, 256), 256, 0, stream>>>(ggc_W + 1 * HID * HID, bt_ggc1, HID, HID, HPB, KPH, 1);
  pack_bt<<<cdiv_i((long)HPB * KPH, 256), 256, 0, stream>>>(ggc_W + 2 * HID * HID, bt_ggc2, HID, HID, HPB, KPH, 1);
  pack_bt<<<cdiv_i((long)H3PB * KPH, 256), 256, 0, stream>>>(gru_wih, bt_wih, HID3, HID, H3PB, KPH, 0);
  pack_bt<<<cdiv_i((long)H3PB * KPH, 256), 256, 0, stream>>>(gru_whh, bt_whh, HID3, HID, H3PB, KPH, 0);
  pack_bt<<<cdiv_i((long)HPB * KP2, 256), 256, 0, stream>>>(fc1_W, bt_fc1, HID, K2, HPB, KP2, 0);

  const int MG   = NNODES / 32;          // 3125 row groups (32 rows each)
  const int NG_H  = HPB / 64;            // 4 n-groups (64 cols each)
  const int NG_H3 = H3PB / 64;           // 10

  // ---- projection: h = relu(bn1(x @ projW^T + b)) ----
  pack_a<<<cdiv_i((long)NNODES * KPH, 256), 256, 0, stream>>>(x, ax_aggb, NNODES, IN_DIM, KPH);
  wmma_gemm_blk<<<cdiv_i((long)MG * NG_H * 32, 256), 256, 0, stream>>>(ax_aggb, bt_proj, R, MG, NG_H, KPH, HP);
  epi_bn_relu_pack<<<cdiv_i((long)NNODES * KPH, 256), 256, 0, stream>>>(
      R, proj_b, bn1_g, bn1_b, bn1_m, bn1_v, h, h_bf, NNODES);

  // ---- 3 GatedGraphConv steps ----
  const unsigned short* bt_ggc[3] = {bt_ggc0, bt_ggc1, bt_ggc2};
  for (int i = 0; i < 3; ++i) {
    // m = h @ ggc_W[i]  -> R
    wmma_gemm_blk<<<cdiv_i((long)MG * NG_H * 32, 256), 256, 0, stream>>>(h_bf, bt_ggc[i], R, MG, NG_H, KPH, HP);
    // agg = segment_sum(m[src], dst)
    hipMemsetAsync(agg, 0, (size_t)NNODES * HID * 4, stream);
    scatter_add<<<cdiv_i((long)NEDGES * 50, 256), 256, 0, stream>>>(R, e_src, e_dst, agg, NEDGES * 50);
    pack_a<<<cdiv_i((long)NNODES * KPH, 256), 256, 0, stream>>>(agg, ax_aggb, NNODES, HID, KPH);
    // gi = agg @ wih^T, gh = h @ whh^T
    wmma_gemm_blk<<<cdiv_i((long)MG * NG_H3 * 32, 256), 256, 0, stream>>>(ax_aggb, bt_wih, Cgi, MG, NG_H3, KPH, H3P);
    wmma_gemm_blk<<<cdiv_i((long)MG * NG_H3 * 32, 256), 256, 0, stream>>>(h_bf, bt_whh, R, MG, NG_H3, KPH, H3P);
    gru_elementwise<<<cdiv_i((long)NNODES * KPH, 256), 256, 0, stream>>>(
        Cgi, R, gru_bih, gru_bhh, h, h_bf, NNODES);
  }

  // ---- BN2 + ReLU (in place) ----
  bn_relu_inplace<<<cdiv_i((long)NNODES * HID, 256), 256, 0, stream>>>(h, bn2_g, bn2_b, bn2_m, bn2_v, NNODES);

  // ---- dual pooling per graph ----
  hipMemsetAsync(p_sum, 0, (size_t)NGRAPHS * HID * 4, stream);
  hipMemsetAsync(p_cnt, 0, (size_t)NGRAPHS * 4, stream);
  hipMemsetAsync(p_max, 0, (size_t)NGRAPHS * HID * 4, stream);  // 0 bits == +0.0f
  pool_count<<<cdiv_i(NNODES, 256), 256, 0, stream>>>(batch, p_cnt, NNODES);
  pool_main<<<cdiv_i((long)NNODES * HID, 256), 256, 0, stream>>>(h, batch, p_sum, p_max, NNODES);
  pool_finalize<<<cdiv_i((long)NGRAPHS * KP2, 256), 256, 0, stream>>>(p_sum, p_max, p_cnt, g_bf);

  // ---- fc1 (WMMA) + BNf + ReLU, then fc2 ----
  wmma_gemm_blk<<<cdiv_i((long)(NGRAPHS / 32) * NG_H * 32, 256), 256, 0, stream>>>(
      g_bf, bt_fc1, Cfc1, NGRAPHS / 32, NG_H, KP2, HP);
  fc1_epilogue<<<cdiv_i((long)NGRAPHS * HID, 256), 256, 0, stream>>>(
      Cfc1, fc1_b, bnf_g, bnf_b, bnf_m, bnf_v, GG);
  fc2_kernel<<<cdiv_i((long)NGRAPHS * 2, 256), 256, 0, stream>>>(GG, fc2_W, fc2_b, out);
}
